// MetaPathGNN_20830591386265
// MI455X (gfx1250) — matrix operand order repaired and checked by reading, not compiled, not run on hardware
//
#include <hip/hip_runtime.h>
#include <hip/hip_bf16.h>

typedef __attribute__((ext_vector_type(16))) __bf16 v16bf;
typedef __attribute__((ext_vector_type(8)))  __bf16 v8bf;
typedef __attribute__((ext_vector_type(8)))  float  v8f;

#define HDIM 256
#define LN_EPS 1e-5f

// ---------- WMMA fragment loads (bf16, 16x16x32) ----------
// A matrix 16x32 (MxK): lane<16 holds K=[k,k+8)+[k+16,k+24); lane>=16 holds
// K=[k+8,k+16)+[k+24,k+32)  (CDNA5 ISA "16-bit A-Matrix 16x32")
__device__ __forceinline__ v16bf load_a_frag(const __bf16* rowPtr, int k, int half) {
  const __bf16* p = rowPtr + k + half * 8;
  v8bf lo = *(const v8bf*)(p);
  v8bf hi = *(const v8bf*)(p + 16);
  return __builtin_shufflevector(lo, hi, 0,1,2,3,4,5,6,7,8,9,10,11,12,13,14,15);
}
// B matrix 32x16 (KxN): lane indexes N; same K order as A: lane<16 holds
// K=[k,k+8)+[k+16,k+24), lane>=16 holds K=[k+8,k+16)+[k+24,k+32). Column j of B == row j of the
// row-major weight (z = h @ W^T) -> two 16B loads.
__device__ __forceinline__ v16bf load_b_frag(const __bf16* colPtr, int k, int half) {
  const __bf16* p = colPtr + k + half * 8;
  v8bf lo = *(const v8bf*)(p);
  v8bf hi = *(const v8bf*)(p + 16);
  return __builtin_shufflevector(lo, hi, 0,1,2,3,4,5,6,7,8,9,10,11,12,13,14,15);
}

// One K=256 GEMM segment, one A stream vs two weight-column blocks.
// Register-level A reuse: 1 A load feeds 2 WMMAs.
__device__ __forceinline__ void gemm_seg(const __bf16* __restrict__ arow,
                                         const __bf16* __restrict__ wrow0,
                                         const __bf16* __restrict__ wrow1,
                                         int half, v8f& acc0, v8f& acc1) {
  #pragma unroll
  for (int k = 0; k < HDIM; k += 32) {
    v16bf a  = load_a_frag(arow, k, half);
    v16bf b0 = load_b_frag(wrow0, k, half);
    v16bf b1 = load_b_frag(wrow1, k, half);
    acc0 = __builtin_amdgcn_wmma_f32_16x16x32_bf16(false, a, false, b0, (short)0, acc0, false, false);
    acc1 = __builtin_amdgcn_wmma_f32_16x16x32_bf16(false, a, false, b1, (short)0, acc1, false, false);
  }
}

// ---------- f32 -> bf16 conversion ----------
__global__ void f2bf_kernel(const float* __restrict__ in, __bf16* __restrict__ out, long n) {
  long i = (long)blockIdx.x * blockDim.x + threadIdx.x;
  long stride = (long)gridDim.x * blockDim.x;
  for (; i < n; i += stride) out[i] = (__bf16)in[i];
}

// ---------- edge scatter-add: agg[e0] += h[e1] ----------
// one block per edge, one feature per thread; f32 atomic accumulation
__global__ __launch_bounds__(HDIM) void scatter_kernel(const __bf16* __restrict__ src,
                                                       const int* __restrict__ e0,
                                                       const int* __restrict__ e1,
                                                       float* __restrict__ aggf) {
  int e = blockIdx.x;
  int f = threadIdx.x;
  int s = e1[e];
  int d = e0[e];
  float v = (float)src[(size_t)s * HDIM + f];
  atomicAdd(&aggf[(size_t)d * HDIM + f], v);
}

// ---------- fused 3x GEMM + bias + ReLU + LayerNorm ----------
// block = 256 threads = 8 waves; wave w owns 2 column blocks:
// cols [32w,32w+16) and [32w+16,32w+32). blockIdx.x owns rows [16bx,16bx+16).
__global__ __launch_bounds__(256)
void gnn_layer_kernel(const __bf16* __restrict__ aggb,
                      const __bf16* __restrict__ hb,
                      const __bf16* __restrict__ xb,
                      const __bf16* __restrict__ Wl,
                      const __bf16* __restrict__ W0,
                      const __bf16* __restrict__ W1,
                      const float* __restrict__ bl,
                      const float* __restrict__ b0,
                      const float* __restrict__ b1,
                      const float* __restrict__ gamma,
                      const float* __restrict__ beta,
                      __bf16* __restrict__ out,
                      int Nn) {
  __shared__ float s_sum[16][8];
  __shared__ float s_sq[16][8];
  __shared__ float s_mean[16];
  __shared__ float s_rstd[16];

  const int w    = threadIdx.x >> 5;   // wave id 0..7
  const int lane = threadIdx.x & 31;
  const int half = lane >> 4;
  const int l15  = lane & 15;
  const int m0   = blockIdx.x * 16;
  const int j0   = w * 32 + l15;       // first column block
  const int j1   = j0 + 16;            // second column block

  int mrow = m0 + l15; if (mrow >= Nn) mrow = Nn - 1;
  const __bf16* arow = aggb + (size_t)mrow * HDIM;
  const __bf16* hrow = hb   + (size_t)mrow * HDIM;
  const __bf16* xrow = xb   + (size_t)mrow * HDIM;

  v8f acc0 = {}, acc1 = {};
  gemm_seg(arow, Wl + (size_t)j0 * HDIM, Wl + (size_t)j1 * HDIM, half, acc0, acc1);
  gemm_seg(hrow, W0 + (size_t)j0 * HDIM, W0 + (size_t)j1 * HDIM, half, acc0, acc1);
  gemm_seg(xrow, W1 + (size_t)j0 * HDIM, W1 + (size_t)j1 * HDIM, half, acc0, acc1);

  // bias (column only) + ReLU.  C/D layout: row M = half*8 + vgpr, col N = l15.
  const float bias0 = bl[j0] + b0[j0] + b1[j0];
  const float bias1 = bl[j1] + b0[j1] + b1[j1];
  float vals0[8], vals1[8];
  #pragma unroll
  for (int r = 0; r < 8; ++r) {
    vals0[r] = fmaxf(acc0[r] + bias0, 0.0f);
    vals1[r] = fmaxf(acc1[r] + bias1, 0.0f);
  }

  // per-row stats: both column blocks feed the same rows, so fold them before
  // the 16-lane shuffle reduce; then LDS reduce across the 8 waves.
  #pragma unroll
  for (int r = 0; r < 8; ++r) {
    float s = vals0[r] + vals1[r];
    float q = vals0[r] * vals0[r] + vals1[r] * vals1[r];
    #pragma unroll
    for (int off = 1; off < 16; off <<= 1) {
      s += __shfl_xor(s, off, 32);
      q += __shfl_xor(q, off, 32);
    }
    if (l15 == 0) {
      int row = half * 8 + r;
      s_sum[row][w] = s;
      s_sq[row][w]  = q;
    }
  }
  __syncthreads();
  if (threadIdx.x < 16) {
    float s = 0.0f, q = 0.0f;
    #pragma unroll
    for (int ww = 0; ww < 8; ++ww) { s += s_sum[threadIdx.x][ww]; q += s_sq[threadIdx.x][ww]; }
    float mean = s * (1.0f / HDIM);
    float var  = q * (1.0f / HDIM) - mean * mean;
    s_mean[threadIdx.x] = mean;
    s_rstd[threadIdx.x] = rsqrtf(var + LN_EPS);
  }
  __syncthreads();

  const float gm0 = gamma[j0], bt0 = beta[j0];
  const float gm1 = gamma[j1], bt1 = beta[j1];
  #pragma unroll
  for (int r = 0; r < 8; ++r) {
    int m = half * 8 + r;
    if (m0 + m < Nn) {
      float rs = s_rstd[m], mu = s_mean[m];
      out[(size_t)(m0 + m) * HDIM + j0] = (__bf16)((vals0[r] - mu) * rs * gm0 + bt0);
      out[(size_t)(m0 + m) * HDIM + j1] = (__bf16)((vals1[r] - mu) * rs * gm1 + bt1);
    }
  }
}

// ---------- readout GEMM: out[N,128] = h @ Wout^T + bout (f32 out) ----------
// 128 threads = 4 waves x 2 column blocks = 128 output columns.
__global__ __launch_bounds__(128)
void out_gemm_kernel(const __bf16* __restrict__ hb,
                     const __bf16* __restrict__ Woutb,
                     const float* __restrict__ bout,
                     float* __restrict__ out,
                     int Nn, int OUTD) {
  const int w    = threadIdx.x >> 5;   // 0..3
  const int lane = threadIdx.x & 31;
  const int half = lane >> 4;
  const int l15  = lane & 15;
  const int m0   = blockIdx.x * 16;
  const int j0   = w * 32 + l15;
  const int j1   = j0 + 16;

  int mrow = m0 + l15; if (mrow >= Nn) mrow = Nn - 1;
  const __bf16* arow = hb + (size_t)mrow * HDIM;

  v8f acc0 = {}, acc1 = {};
  gemm_seg(arow, Woutb + (size_t)j0 * HDIM, Woutb + (size_t)j1 * HDIM, half, acc0, acc1);

  const float bj0 = bout[j0];
  const float bj1 = bout[j1];
  #pragma unroll
  for (int r = 0; r < 8; ++r) {
    int m = half * 8 + r;
    if (m0 + m < Nn) {
      out[(size_t)(m0 + m) * OUTD + j0] = acc0[r] + bj0;
      out[(size_t)(m0 + m) * OUTD + j1] = acc1[r] + bj1;
    }
  }
}

extern "C" void kernel_launch(void* const* d_in, const int* in_sizes, int n_in,
                              void* d_out, int out_size, void* d_ws, size_t ws_size,
                              hipStream_t stream) {
  const float* x     = (const float*)d_in[0];
  const int*   er1   = (const int*)d_in[1];
  const int*   er2   = (const int*)d_in[2];
  const float* Wl    = (const float*)d_in[3];
  const float* bl    = (const float*)d_in[4];
  const float* W0    = (const float*)d_in[5];
  const float* b0    = (const float*)d_in[6];
  const float* W1    = (const float*)d_in[7];
  const float* b1    = (const float*)d_in[8];
  const float* gamma = (const float*)d_in[9];
  const float* beta  = (const float*)d_in[10];
  const float* Wout  = (const float*)d_in[11];
  const float* bout  = (const float*)d_in[12];
  float* out = (float*)d_out;

  const int Nn   = in_sizes[0] / HDIM;   // 50000
  const int E1   = in_sizes[1] / 2;      // 500000
  const int E2   = in_sizes[2] / 2;
  const int OUTD = in_sizes[11] / HDIM;  // 128
  const long nWl = in_sizes[3];          // L*H*H
  const long nWo = in_sizes[11];

  // workspace carve-out (256B aligned)
  char* p = (char*)d_ws;
  auto alloc = [&](size_t bytes) -> char* {
    char* q = p;
    p += (bytes + 255) & ~(size_t)255;
    return q;
  };
  const size_t nhb = (size_t)Nn * HDIM;
  __bf16* xb    = (__bf16*)alloc(nhb * 2);
  __bf16* hb1   = (__bf16*)alloc(nhb * 2);
  __bf16* hb2   = (__bf16*)alloc(nhb * 2);
  __bf16* aggb  = (__bf16*)alloc(nhb * 2);
  float*  aggf  = (float*)alloc(nhb * 4);
  __bf16* Wlb   = (__bf16*)alloc((size_t)nWl * 2);
  __bf16* W0b   = (__bf16*)alloc((size_t)nWl * 2);
  __bf16* W1b   = (__bf16*)alloc((size_t)nWl * 2);
  __bf16* Woutb = (__bf16*)alloc((size_t)nWo * 2);

  const int cvtBlocks = 2048;
  // one-time conversions
  f2bf_kernel<<<cvtBlocks, 256, 0, stream>>>(x, xb, (long)nhb);
  f2bf_kernel<<<cvtBlocks, 256, 0, stream>>>(Wl, Wlb, nWl);
  f2bf_kernel<<<cvtBlocks, 256, 0, stream>>>(W0, W0b, nWl);
  f2bf_kernel<<<cvtBlocks, 256, 0, stream>>>(W1, W1b, nWl);
  f2bf_kernel<<<256, 256, 0, stream>>>(Wout, Woutb, nWo);

  const int tiles = (Nn + 15) / 16;
  const size_t layerOff = (size_t)HDIM * HDIM;  // weight layer stride

  // ----- layer 1 of reversed metapath: conv_idx=1, edges = edge_r2, h = x -----
  hipMemsetAsync(aggf, 0, nhb * sizeof(float), stream);
  scatter_kernel<<<E2, HDIM, 0, stream>>>(xb, er2, er2 + E2, aggf);
  f2bf_kernel<<<cvtBlocks, 256, 0, stream>>>(aggf, aggb, (long)nhb);
  gnn_layer_kernel<<<tiles, 256, 0, stream>>>(
      aggb, xb, xb,
      Wlb + layerOff, W0b + layerOff, W1b + layerOff,
      bl + HDIM, b0 + HDIM, b1 + HDIM, gamma + HDIM, beta + HDIM,
      hb1, Nn);

  // ----- layer 2: conv_idx=0, edges = edge_r1, h = hb1 -----
  hipMemsetAsync(aggf, 0, nhb * sizeof(float), stream);
  scatter_kernel<<<E1, HDIM, 0, stream>>>(hb1, er1, er1 + E1, aggf);
  f2bf_kernel<<<cvtBlocks, 256, 0, stream>>>(aggf, aggb, (long)nhb);
  gnn_layer_kernel<<<tiles, 256, 0, stream>>>(
      aggb, hb1, xb,
      Wlb, W0b, W1b,
      bl, b0, b1, gamma, beta,
      hb2, Nn);

  // ----- readout -----
  out_gemm_kernel<<<tiles, 128, 0, stream>>>(hb2, Woutb, bout, out, Nn, OUTD);
}
